// ae_simclr_68358699483161
// MI455X (gfx1250) — matrix operand order repaired and checked by reading, not compiled, not run on hardware
//
#include <hip/hip_runtime.h>
#include <hip/hip_bf16.h>

#define B_    256
#define W_    128
#define CIN_  128
#define H_    512
#define CODE_ 256
#define OUT_  64
#define H4_   2048
#define LRELU_ 0.2f
#define BN_EPS_ 1e-5f
#define BSTRIDE_ 72   // padded LDS row stride (bf16 elems): 144B, kills bank conflicts

typedef __attribute__((ext_vector_type(16))) __bf16 v16bf;
typedef __attribute__((ext_vector_type(8)))  float  v8f;

union FragBF { v16bf v; uint4 q[2]; };

__device__ __forceinline__ float sigmoidf_(float x) { return 1.0f / (1.0f + __expf(-x)); }

// gfx1250 async global->LDS DMA (ASYNCcnt); INST_OFFSET applies to both sides.
#define ASYNC_B128(ldsaddr, gaddr, OFF)                                            \
  asm volatile("global_load_async_to_lds_b128 %0, %1, off offset:" #OFF            \
               :: "v"(ldsaddr), "v"(gaddr) : "memory")
#define WAIT_ASYNC() asm volatile("s_wait_asynccnt 0" ::: "memory")

// C[m][n] = sum_k A[m][k] * Bw[n][k] + bias[n]
// A split at ksplit between A1/A2 (concatenated K); split is uniform per 64-chunk.
// Bw row-major [N][Ktot]. Block = 128 thr = 4 waves; wave w: rows m0+16w..+15,
// cols n0..n0+63. grid = (N/64, M/64). M,N mult of 64; Ktot mult of 64; ksplit mult of 64.
// B chunk (64x64 bf16) staged in LDS via async DMA, double-buffered.
// rm!=0: output row remap  row = (m % B_)*W_ + m/B_   ([t][b] -> [b][t]); pow-2 consts.
__global__ __launch_bounds__(128) void gemm_wmma_bf16(
    const __hip_bfloat16* __restrict__ A1, int lda1,
    const __hip_bfloat16* __restrict__ A2, int lda2, int ksplit,
    const __hip_bfloat16* __restrict__ Bw, int ldb,
    const float* __restrict__ bias,
    float* __restrict__ C, int ldc,
    int Ktot, int rm) {
  __shared__ __hip_bfloat16 Bsh[2][64 * BSTRIDE_];

  const int tid  = threadIdx.x;
  const int lane = tid & 31;
  const int wave = tid >> 5;
  const int r    = lane & 15;
  const int half = lane >> 4;
  const int m0 = blockIdx.y * 64 + wave * 16;
  const int n0 = blockIdx.x * 64;
  const int m  = m0 + r;

  // cooperative stage: thread copies 64B: row (n0 + tid/2), cols (tid&1)*32 .. +31
  const int brow = tid >> 1;
  const int bcol = (tid & 1) * 32;
  const char* gB = (const char*)(Bw + (size_t)(n0 + brow) * ldb + bcol);
  const unsigned lds0 = (unsigned)(uintptr_t)&Bsh[0][brow * BSTRIDE_ + bcol];
  const unsigned lds1 = (unsigned)(uintptr_t)&Bsh[1][brow * BSTRIDE_ + bcol];

  v8f acc0 = {}, acc1 = {}, acc2 = {}, acc3 = {};
  const int nchunks = Ktot >> 6;

  // prologue: DMA chunk 0 into buffer 0
  ASYNC_B128(lds0, gB, 0);
  ASYNC_B128(lds0, gB, 16);
  ASYNC_B128(lds0, gB, 32);
  ASYNC_B128(lds0, gB, 48);
  WAIT_ASYNC();
  __syncthreads();

  for (int c = 0; c < nchunks; ++c) {
    const int k0 = c << 6;
    const bool more = (c + 1) < nchunks;
    if (more) {  // DMA next chunk into the other buffer while computing
      const char* g = gB + (size_t)(k0 + 64) * 2;
      const unsigned l = ((c + 1) & 1) ? lds1 : lds0;
      ASYNC_B128(l, g, 0);
      ASYNC_B128(l, g, 16);
      ASYNC_B128(l, g, 32);
      ASYNC_B128(l, g, 48);
    }

    // A base for this chunk: uniform source select (no per-lane cndmask)
    const __hip_bfloat16* arow;
    if (k0 < ksplit) arow = A1 + (size_t)m * lda1 + k0;
    else             arow = A2 + (size_t)m * lda2 + (k0 - ksplit);

    const __hip_bfloat16* bsh = &Bsh[c & 1][0];
#pragma unroll
    for (int kk = 0; kk < 64; kk += 32) {
      // A frag: lane (half,r): K = kk + 16*half .. +15 contiguous (matches B).
      FragBF a;
      a.q[0] = *reinterpret_cast<const uint4*>(arow + kk + 16 * half);
      a.q[1] = *reinterpret_cast<const uint4*>(arow + kk + 16 * half + 8);
      // B frag: lane (half,r): N row = 16j + r, K = kk + 16*half .. +15 contiguous.
      const __hip_bfloat16* bb = bsh + kk + 16 * half;
      FragBF b0, b1, b2, b3;
      { const uint4* p = reinterpret_cast<const uint4*>(bb + ( 0 + r) * BSTRIDE_); b0.q[0] = p[0]; b0.q[1] = p[1]; }
      { const uint4* p = reinterpret_cast<const uint4*>(bb + (16 + r) * BSTRIDE_); b1.q[0] = p[0]; b1.q[1] = p[1]; }
      { const uint4* p = reinterpret_cast<const uint4*>(bb + (32 + r) * BSTRIDE_); b2.q[0] = p[0]; b2.q[1] = p[1]; }
      { const uint4* p = reinterpret_cast<const uint4*>(bb + (48 + r) * BSTRIDE_); b3.q[0] = p[0]; b3.q[1] = p[1]; }

      acc0 = __builtin_amdgcn_wmma_f32_16x16x32_bf16(false, a.v, false, b0.v, (short)0, acc0, false, false);
      acc1 = __builtin_amdgcn_wmma_f32_16x16x32_bf16(false, a.v, false, b1.v, (short)0, acc1, false, false);
      acc2 = __builtin_amdgcn_wmma_f32_16x16x32_bf16(false, a.v, false, b2.v, (short)0, acc2, false, false);
      acc3 = __builtin_amdgcn_wmma_f32_16x16x32_bf16(false, a.v, false, b3.v, (short)0, acc3, false, false);
    }

    if (more) {
      WAIT_ASYNC();     // my DMA done; barrier ensures everyone's is
      __syncthreads();
    }
  }

  // Epilogue: C/D 16x16 layout — lane L: n = L%16, VGPR v: m = v + 8*(L/16).
  // Row remap uses compile-time pow-2 constants (mask/shift, no integer division).
#pragma unroll
  for (int v = 0; v < 8; ++v) {
    const int mr = m0 + v + 8 * half;
    const int row = rm ? ((mr & (B_ - 1)) * W_ + (mr >> 8)) : mr;
    float* crow = C + (size_t)row * ldc;
    int n;
    n = n0 +  0 + r; crow[n] = acc0[v] + (bias ? bias[n] : 0.0f);
    n = n0 + 16 + r; crow[n] = acc1[v] + (bias ? bias[n] : 0.0f);
    n = n0 + 32 + r; crow[n] = acc2[v] + (bias ? bias[n] : 0.0f);
    n = n0 + 48 + r; crow[n] = acc3[v] + (bias ? bias[n] : 0.0f);
  }
}

// ---- elementwise / reduction helpers ----

__global__ void k_cvt_bf16(const float* __restrict__ s, __hip_bfloat16* __restrict__ d,
                           int n, int leaky) {
  int i = blockIdx.x * 256 + threadIdx.x;
  if (i >= n) return;
  float v = s[i];
  if (leaky) v = v > 0.0f ? v : LRELU_ * v;
  d[i] = __float2bfloat16(v);
}

__global__ void k_concat_bf16(const float* __restrict__ Wih, const float* __restrict__ Whh,
                              __hip_bfloat16* __restrict__ d, int N, int K) {
  int i = blockIdx.x * 256 + threadIdx.x;
  if (i >= N * 2 * K) return;
  int n = i / (2 * K), k = i % (2 * K);
  float v = (k < K) ? Wih[(size_t)n * K + k] : Whh[(size_t)n * K + (k - K)];
  d[i] = __float2bfloat16(v);
}

__global__ void k_addcvt_bf16(const float* __restrict__ a, const float* __restrict__ b,
                              __hip_bfloat16* __restrict__ d, int n) {
  int i = blockIdx.x * 256 + threadIdx.x;
  if (i >= n) return;
  d[i] = __float2bfloat16(a[i] + b[i]);
}

__global__ void k_vadd(const float* __restrict__ a, const float* __restrict__ b,
                       float* __restrict__ d, int n) {
  int i = blockIdx.x * 256 + threadIdx.x;
  if (i >= n) return;
  d[i] = a[i] + b[i];
}

__global__ void k_add2out(const float* __restrict__ a, const float* __restrict__ b,
                          float* __restrict__ df, __hip_bfloat16* __restrict__ db, int n) {
  int i = blockIdx.x * 256 + threadIdx.x;
  if (i >= n) return;
  float v = a[i] + b[i];
  df[i] = v;
  db[i] = __float2bfloat16(v);
}

__global__ void k_zero(float* __restrict__ p, int n) {
  int i = blockIdx.x * 256 + threadIdx.x;
  if (i < n) p[i] = 0.0f;
}

__global__ void k_bn_stats(const float* __restrict__ xf, const float* __restrict__ gamma,
                           const float* __restrict__ beta, float* __restrict__ scale,
                           float* __restrict__ shift, int rows) {
  int c = blockIdx.x * blockDim.x + threadIdx.x;
  if (c >= H_) return;
  float s = 0.0f, ss = 0.0f;
  for (int r0 = 0; r0 < rows; ++r0) {
    float v = xf[(size_t)r0 * H_ + c];
    s += v; ss += v * v;
  }
  float inv = 1.0f / (float)rows;
  float mean = s * inv;
  float var  = ss * inv - mean * mean;
  float sc = gamma[c] * rsqrtf(var + BN_EPS_);
  scale[c] = sc;
  shift[c] = beta[c] - mean * sc;
}

// normalize+leaky, reorder rows [b*W+w] -> [w*B+b], emit bf16 for WMMA A use.
__global__ void k_bn_apply(const float* __restrict__ xf, const float* __restrict__ scale,
                           const float* __restrict__ shift, __hip_bfloat16* __restrict__ xe) {
  int i = blockIdx.x * 256 + threadIdx.x;
  if (i >= B_ * W_ * H_) return;
  int c = i & (H_ - 1);
  int row = i >> 9;            // b*W + w
  int b = row >> 7, w = row & (W_ - 1);
  float v = xf[i] * scale[c] + shift[c];
  v = v > 0.0f ? v : LRELU_ * v;
  xe[((size_t)(w * B_ + b)) * H_ + c] = __float2bfloat16(v);
}

// LSTM cell: gates [B_][4H] (i,f,g,o); updates h,c (f32) and writes bf16 h.
__global__ void k_lstm_cell(const float* __restrict__ gates, float* __restrict__ h,
                            float* __restrict__ c, __hip_bfloat16* __restrict__ h_bf) {
  int i = blockIdx.x * 256 + threadIdx.x;
  if (i >= B_ * H_) return;
  int b = i >> 9, j = i & (H_ - 1);
  const float* g = gates + (size_t)b * H4_;
  float gi = g[j], gf = g[H_ + j], gg = g[2 * H_ + j], go = g[3 * H_ + j];
  float cn = sigmoidf_(gf) * c[i] + sigmoidf_(gi) * tanhf(gg);
  c[i] = cn;
  float hn = sigmoidf_(go) * tanhf(cn);
  h[i] = hn;
  h_bf[i] = __float2bfloat16(hn);
}

// ---- orchestration ----

static inline void gemm(hipStream_t st, const __hip_bfloat16* A1, int lda1,
                        const __hip_bfloat16* A2, int lda2, int ksplit,
                        const __hip_bfloat16* Bw, int ldb, const float* bias,
                        float* C, int ldc, int M, int N, int Ktot, int rm = 0) {
  dim3 g(N / 64, M / 64), blk(128);
  gemm_wmma_bf16<<<g, blk, 0, st>>>(A1, lda1, A2, lda2, ksplit, Bw, ldb, bias, C, ldc,
                                    Ktot, rm);
}
static inline int nb(int n) { return (n + 255) / 256; }

extern "C" void kernel_launch(void* const* d_in, const int* in_sizes, int n_in,
                              void* d_out, int out_size, void* d_ws, size_t ws_size,
                              hipStream_t stream) {
  (void)in_sizes; (void)n_in; (void)out_size; (void)ws_size;
  const float* x        = (const float*)d_in[0];
  const float* enc_W    = (const float*)d_in[1];
  const float* enc_b    = (const float*)d_in[2];
  const float* gamma    = (const float*)d_in[3];
  const float* beta     = (const float*)d_in[4];
  const float* Wih_f    = (const float*)d_in[5];
  const float* Whh_f    = (const float*)d_in[6];
  const float* bih_f    = (const float*)d_in[7];
  const float* bhh_f    = (const float*)d_in[8];
  const float* Wih_b    = (const float*)d_in[9];
  const float* Whh_b    = (const float*)d_in[10];
  const float* bih_b    = (const float*)d_in[11];
  const float* bhh_b    = (const float*)d_in[12];
  const float* code_W   = (const float*)d_in[13];
  const float* code_b   = (const float*)d_in[14];
  const float* sim_W    = (const float*)d_in[15];
  const float* sim_b    = (const float*)d_in[16];
  const float* static_W = (const float*)d_in[17];
  const float* static_b = (const float*)d_in[18];
  const float* dec_W    = (const float*)d_in[19];
  const float* dec_b    = (const float*)d_in[20];
  const float* Wih_d    = (const float*)d_in[21];
  const float* Whh_d    = (const float*)d_in[22];
  const float* bih_d    = (const float*)d_in[23];
  const float* bhh_d    = (const float*)d_in[24];
  const float* inst_W   = (const float*)d_in[25];
  const float* inst_b   = (const float*)d_in[26];

  float* out      = (float*)d_out;
  float* out_z    = out;                          // [B,CODE]
  float* out_enc  = out + 65536;                  // [B,H]
  float* out_sim  = out + 196608;                 // [B,CODE]
  float* out_stat = out + 262144;                 // [B,OUT]
  float* out_dec  = out + 278528;                 // [B,W,CIN]

  char* ws = (char*)d_ws;
  size_t off = 0;
  auto alloc = [&](size_t bytes) -> char* {
    char* p = ws + off;
    off = (off + bytes + 255) & ~(size_t)255;
    return p;
  };
  const size_t BH = (size_t)B_ * H_;

  __hip_bfloat16* x_bf    = (__hip_bfloat16*)alloc((size_t)B_ * W_ * CIN_ * 2);
  float*          xf      = (float*)alloc((size_t)B_ * W_ * H_ * 4);
  __hip_bfloat16* xe      = (__hip_bfloat16*)alloc((size_t)B_ * W_ * H_ * 2);
  __hip_bfloat16* encW_bf = (__hip_bfloat16*)alloc((size_t)H_ * CIN_ * 2);
  __hip_bfloat16* Wcat_f  = (__hip_bfloat16*)alloc((size_t)H4_ * 2 * H_ * 2);
  __hip_bfloat16* Wcat_b  = (__hip_bfloat16*)alloc((size_t)H4_ * 2 * H_ * 2);
  __hip_bfloat16* Whhd_bf = (__hip_bfloat16*)alloc((size_t)H4_ * H_ * 2);
  __hip_bfloat16* Wsum_bf = (__hip_bfloat16*)alloc((size_t)H4_ * H_ * 2);
  __hip_bfloat16* code_bf = (__hip_bfloat16*)alloc((size_t)CODE_ * H_ * 2);
  __hip_bfloat16* sim_bf  = (__hip_bfloat16*)alloc((size_t)CODE_ * CODE_ * 2);
  __hip_bfloat16* stat_bf = (__hip_bfloat16*)alloc((size_t)OUT_ * CODE_ * 2);
  __hip_bfloat16* decW_bf = (__hip_bfloat16*)alloc((size_t)H_ * CODE_ * 2);
  __hip_bfloat16* inst_bf = (__hip_bfloat16*)alloc((size_t)CIN_ * H_ * 2);
  float* bias_f = (float*)alloc(H4_ * 4);
  float* bias_b = (float*)alloc(H4_ * 4);
  float* bias_d = (float*)alloc(H4_ * 4);
  float* scale  = (float*)alloc(H_ * 4);
  float* shift  = (float*)alloc(H_ * 4);
  float* gates  = (float*)alloc((size_t)B_ * H4_ * 4);
  float* h_f = (float*)alloc(BH * 4);
  float* c_f = (float*)alloc(BH * 4);
  float* h_b = (float*)alloc(BH * 4);
  float* c_b = (float*)alloc(BH * 4);
  float* c_d = (float*)alloc(BH * 4);
  float* dec_vec = (float*)alloc(BH * 4);         // decoder h (f32)
  __hip_bfloat16* hf_bf  = (__hip_bfloat16*)alloc(BH * 2);
  __hip_bfloat16* hb_bf  = (__hip_bfloat16*)alloc(BH * 2);
  __hip_bfloat16* enc_bf = (__hip_bfloat16*)alloc(BH * 2);
  __hip_bfloat16* dv_bf  = (__hip_bfloat16*)alloc(BH * 2);
  __hip_bfloat16* z_bf   = (__hip_bfloat16*)alloc((size_t)B_ * CODE_ * 2);
  __hip_bfloat16* zlr_bf = (__hip_bfloat16*)alloc((size_t)B_ * CODE_ * 2);
  __hip_bfloat16* hdall  = (__hip_bfloat16*)alloc((size_t)W_ * B_ * H_ * 2);

  // ---- weight / bias prep ----
  int n;
  n = B_ * W_ * CIN_; k_cvt_bf16<<<nb(n), 256, 0, stream>>>(x, x_bf, n, 0);
  n = H_ * CIN_;      k_cvt_bf16<<<nb(n), 256, 0, stream>>>(enc_W, encW_bf, n, 0);
  n = H4_ * 2 * H_;   k_concat_bf16<<<nb(n), 256, 0, stream>>>(Wih_f, Whh_f, Wcat_f, H4_, H_);
                      k_concat_bf16<<<nb(n), 256, 0, stream>>>(Wih_b, Whh_b, Wcat_b, H4_, H_);
  n = H4_ * H_;       k_cvt_bf16<<<nb(n), 256, 0, stream>>>(Whh_d, Whhd_bf, n, 0);
                      k_addcvt_bf16<<<nb(n), 256, 0, stream>>>(Wih_d, Whh_d, Wsum_bf, n);
  n = CODE_ * H_;     k_cvt_bf16<<<nb(n), 256, 0, stream>>>(code_W, code_bf, n, 0);
  n = CODE_ * CODE_;  k_cvt_bf16<<<nb(n), 256, 0, stream>>>(sim_W, sim_bf, n, 0);
  n = OUT_ * CODE_;   k_cvt_bf16<<<nb(n), 256, 0, stream>>>(static_W, stat_bf, n, 0);
  n = H_ * CODE_;     k_cvt_bf16<<<nb(n), 256, 0, stream>>>(dec_W, decW_bf, n, 0);
  n = CIN_ * H_;      k_cvt_bf16<<<nb(n), 256, 0, stream>>>(inst_W, inst_bf, n, 0);
  k_vadd<<<nb(H4_), 256, 0, stream>>>(bih_f, bhh_f, bias_f, H4_);
  k_vadd<<<nb(H4_), 256, 0, stream>>>(bih_b, bhh_b, bias_b, H4_);
  k_vadd<<<nb(H4_), 256, 0, stream>>>(bih_d, bhh_d, bias_d, H4_);

  // ---- instance encoder: linear -> BN(train stats) -> leaky -> [w][b][H] bf16 ----
  gemm(stream, x_bf, CIN_, nullptr, 0, 1 << 30, encW_bf, CIN_, enc_b,
       xf, H_, B_ * W_, H_, CIN_);
  k_bn_stats<<<2, 256, 0, stream>>>(xf, gamma, beta, scale, shift, B_ * W_);
  n = B_ * W_ * H_;
  k_bn_apply<<<nb(n), 256, 0, stream>>>(xf, scale, shift, xe);

  // ---- bidirectional LSTM (fused [x_t|h] @ [Wih|Whh].T per step) ----
  n = (int)BH;
  k_zero<<<nb(n), 256, 0, stream>>>(h_f, n);
  k_zero<<<nb(n), 256, 0, stream>>>(c_f, n);
  k_zero<<<nb(n), 256, 0, stream>>>(h_b, n);
  k_zero<<<nb(n), 256, 0, stream>>>(c_b, n);
  k_zero<<<nb(n / 2), 256, 0, stream>>>((float*)hf_bf, n / 2);
  k_zero<<<nb(n / 2), 256, 0, stream>>>((float*)hb_bf, n / 2);

  for (int t = 0; t < W_; ++t) {
    gemm(stream, xe + (size_t)t * BH, H_, hf_bf, H_, H_, Wcat_f, 2 * H_, bias_f,
         gates, H4_, B_, H4_, 2 * H_);
    k_lstm_cell<<<nb((int)BH), 256, 0, stream>>>(gates, h_f, c_f, hf_bf);
  }
  for (int t = 0; t < W_; ++t) {
    gemm(stream, xe + (size_t)(W_ - 1 - t) * BH, H_, hb_bf, H_, H_, Wcat_b, 2 * H_, bias_b,
         gates, H4_, B_, H4_, 2 * H_);
    k_lstm_cell<<<nb((int)BH), 256, 0, stream>>>(gates, h_b, c_b, hb_bf);
  }
  k_add2out<<<nb((int)BH), 256, 0, stream>>>(h_f, h_b, out_enc, enc_bf, (int)BH);

  // ---- heads ----
  gemm(stream, enc_bf, H_, nullptr, 0, 1 << 30, code_bf, H_, code_b,
       out_z, CODE_, B_, CODE_, H_);
  n = B_ * CODE_;
  k_cvt_bf16<<<nb(n), 256, 0, stream>>>(out_z, z_bf, n, 0);
  k_cvt_bf16<<<nb(n), 256, 0, stream>>>(out_z, zlr_bf, n, 1);
  gemm(stream, zlr_bf, CODE_, nullptr, 0, 1 << 30, sim_bf, CODE_, sim_b,
       out_sim, CODE_, B_, CODE_, CODE_);
  gemm(stream, z_bf, CODE_, nullptr, 0, 1 << 30, stat_bf, CODE_, static_b,
       out_stat, OUT_, B_, OUT_, CODE_);
  gemm(stream, z_bf, CODE_, nullptr, 0, 1 << 30, decW_bf, CODE_, dec_b,
       dec_vec, H_, B_, H_, CODE_);
  n = (int)BH;
  k_cvt_bf16<<<nb(n), 256, 0, stream>>>(dec_vec, dv_bf, n, 0);
  k_zero<<<nb(n), 256, 0, stream>>>(c_d, n);

  // ---- autoregressive decoder: inp==h for t>=1 -> one GEMM/step with Wsum ----
  gemm(stream, dv_bf, H_, nullptr, 0, 1 << 30, Whhd_bf, H_, bias_d,
       gates, H4_, B_, H4_, H_);                                   // t=0 (inp=0)
  k_lstm_cell<<<nb((int)BH), 256, 0, stream>>>(gates, dec_vec, c_d, hdall);
  for (int t = 1; t < W_; ++t) {
    gemm(stream, hdall + (size_t)(t - 1) * BH, H_, nullptr, 0, 1 << 30, Wsum_bf, H_, bias_d,
         gates, H4_, B_, H4_, H_);
    k_lstm_cell<<<nb((int)BH), 256, 0, stream>>>(gates, dec_vec, c_d, hdall + (size_t)t * BH);
  }

  // ---- batched output projection, rows [t][b] remapped to out[b][t][:] ----
  gemm(stream, hdall, H_, nullptr, 0, 1 << 30, inst_bf, H_, inst_b,
       out_dec, CIN_, W_ * B_, CIN_, H_, /*rm=*/1);
}
